// Attention_8143257993917
// MI455X (gfx1250) — matrix operand order repaired and checked by reading, not compiled, not run on hardware
//
#include <hip/hip_runtime.h>
#include <math.h>

// ---------------------------------------------------------------------------
// MI455X (gfx1250) attention: bf16 WMMA (v_wmma_f32_16x16x32_bf16), wave32,
// async global->LDS staging (ASYNCcnt) for K/V in the attention kernel.
// Shapes fixed by the reference: B=4, S=2048, D=1024, H=16, Hd=64.
// ---------------------------------------------------------------------------

typedef __bf16 bf16;
typedef __attribute__((ext_vector_type(16))) __bf16 v16bf;
typedef __attribute__((ext_vector_type(8)))  __bf16 v8bf;
typedef __attribute__((ext_vector_type(8)))  float  v8f;

#define BATCH 4
#define SEQ   2048
#define DIM   1024
#define NH    16
#define HD    64
#define D3    3072

__device__ __forceinline__ v8f wmma_bf16(v16bf a, v16bf b, v8f c) {
  return __builtin_amdgcn_wmma_f32_16x16x32_bf16(false, a, false, b, (short)0, c,
                                                 false, false);
}

// A-fragment (16x32 bf16, row-major source, stride lda elements).
// ISA layout: lane L (L<16) holds row L, K {0..7, 16..23}; lane L+16 holds
// row L, K {8..15, 24..31}.
__device__ __forceinline__ v16bf load_a_frag(const bf16* A, int lda, int m0,
                                             int kb, int lane) {
  int row = m0 + (lane & 15);
  int k0  = kb + ((lane >> 4) << 3);
  const bf16* p = A + (long)row * lda + k0;
  v8bf lo = *(const v8bf*)(p);
  v8bf hi = *(const v8bf*)(p + 16);
  v16bf a;
#pragma unroll
  for (int i = 0; i < 8; ++i) { a[i] = lo[i]; a[i + 8] = hi[i]; }
  return a;
}

// B-fragment (32x16 bf16). Source Bt is [N][K] row-major (column n of B is a
// contiguous row of Bt), stride ldb elements. Layout: lane L holds column
// L&15, K {0..7, 16..23} (L<16) or K {8..15, 24..31} (L>=16), matching A.
__device__ __forceinline__ v16bf load_b_frag(const bf16* Bt, int ldb, int n0,
                                             int kb, int lane) {
  int col = n0 + (lane & 15);
  int k0  = kb + ((lane >> 4) << 3);
  const bf16* p = Bt + (long)col * ldb + k0;
  v8bf lo = *(const v8bf*)(p);
  v8bf hi = *(const v8bf*)(p + 16);
  v16bf b;
#pragma unroll
  for (int i = 0; i < 8; ++i) { b[i] = lo[i]; b[i + 8] = hi[i]; }
  return b;
}

// ---------------------------------------------------------------------------
// f32 -> bf16 conversion
// ---------------------------------------------------------------------------
__global__ __launch_bounds__(256) void cvt_bf16_kernel(const float* __restrict__ in,
                                                       bf16* __restrict__ out, int n) {
  int i = blockIdx.x * 256 + threadIdx.x;
  if (i < n) out[i] = (bf16)in[i];
}

// ---------------------------------------------------------------------------
// QKV projection: qkv[m][n] = sum_k hs[m][k] * qkv_w[n][k] + qkv_b[n]
// 32x64 tile per wave (2 A-frags x 4 B-frags, 8 WMMAs per k-step).
// Scatter into Q[B,H,S,64], K[B,H,S,64], V^T[B,H,64,S] (bf16).
// ---------------------------------------------------------------------------
__global__ __launch_bounds__(256) void qkv_gemm_kernel(
    const bf16* __restrict__ hsb, const bf16* __restrict__ wb,
    const float* __restrict__ bias, bf16* __restrict__ Qb,
    bf16* __restrict__ Kb, bf16* __restrict__ Vt) {
  int t    = blockIdx.x * 8 + (threadIdx.x >> 5);
  int lane = threadIdx.x & 31;
  int tm = t / (D3 / 64);           // 0..255  (M tiles of 32)
  int tn = t % (D3 / 64);           // 0..47   (N tiles of 64)
  int m0 = tm * 32, n0 = tn * 64;

  v8f acc[2][4] = {};
  for (int kb = 0; kb < DIM; kb += 32) {
    v16bf a0 = load_a_frag(hsb, DIM, m0,      kb, lane);
    v16bf a1 = load_a_frag(hsb, DIM, m0 + 16, kb, lane);
#pragma unroll
    for (int j = 0; j < 4; ++j) {
      v16bf bj = load_b_frag(wb, DIM, n0 + 16 * j, kb, lane);
      acc[0][j] = wmma_bf16(a0, bj, acc[0][j]);
      acc[1][j] = wmma_bf16(a1, bj, acc[1][j]);
    }
  }

  int hi = lane >> 4;
  int cl = lane & 15;
#pragma unroll
  for (int j = 0; j < 4; ++j) {
    int n = n0 + 16 * j + cl;
    float bn = bias[n];
    int which = n >> 10;            // 0=Q, 1=K, 2=V  (uniform per tile)
    int hn = (n & 1023) >> 6;       // head           (uniform per tile)
    int dn = n & 63;
#pragma unroll
    for (int i = 0; i < 2; ++i) {
#pragma unroll
      for (int r = 0; r < 8; ++r) {
        int m = m0 + 16 * i + r + 8 * hi;
        int bidx = m >> 11;         // / SEQ
        int s    = m & (SEQ - 1);
        bf16 v = (bf16)(acc[i][j][r] + bn);
        long bh = (long)(bidx * NH + hn);
        if (which == 0)      Qb[(bh * SEQ + s) * HD + dn] = v;
        else if (which == 1) Kb[(bh * SEQ + s) * HD + dn] = v;
        else                 Vt[(bh * HD + dn) * SEQ + s] = v;
      }
    }
  }
}

// ---------------------------------------------------------------------------
// RoPE (in place, bf16 Q/K). Each thread owns the (d, d+32) pair -> no races.
// ---------------------------------------------------------------------------
__global__ __launch_bounds__(256) void rope_kernel(bf16* __restrict__ Qb,
                                                   bf16* __restrict__ Kb,
                                                   const float* __restrict__ cosb,
                                                   const float* __restrict__ sinb) {
  int idx = blockIdx.x * 256 + threadIdx.x;   // B*H*S*32 = 4194304 threads
  int d   = idx & 31;
  int sh  = idx >> 5;                         // (b*H+h)*S + s
  int s   = sh & (SEQ - 1);
  long base = (long)sh * HD;
  float c0 = cosb[s * HD + d],      c1 = cosb[s * HD + d + 32];
  float sn0 = sinb[s * HD + d],     sn1 = sinb[s * HD + d + 32];
  float q0 = (float)Qb[base + d],   q1 = (float)Qb[base + d + 32];
  Qb[base + d]      = (bf16)(q0 * c0 - q1 * sn0);
  Qb[base + d + 32] = (bf16)(q1 * c1 + q0 * sn1);
  float k0 = (float)Kb[base + d],   k1 = (float)Kb[base + d + 32];
  Kb[base + d]      = (bf16)(k0 * c0 - k1 * sn0);
  Kb[base + d + 32] = (bf16)(k1 * c1 + k0 * sn1);
}

// ---------------------------------------------------------------------------
// Async global->LDS staging of one 32-key chunk of K [32][64] and V^T [64][32].
// Each of the 256 threads copies one 16B slice of each (4KB per buffer),
// tracked by ASYNCcnt.
// ---------------------------------------------------------------------------
__device__ __forceinline__ void stage_kv(const bf16* Kh, const bf16* Vh, int kc,
                                         bf16* kdst, bf16* vdst, int t) {
  // K chunk is contiguous in global: rows kc..kc+31, 64 dims each.
  const bf16* kg = Kh + (long)kc * HD + t * 8;
  unsigned kl = (unsigned)(size_t)kdst + (unsigned)t * 16u;
  asm volatile("global_load_async_to_lds_b128 %0, %1, off"
               :: "v"(kl), "v"(kg) : "memory");
  // V^T chunk: 64 rows (dims) x 32 keys, row stride SEQ in global.
  const bf16* vg = Vh + (long)(t >> 2) * SEQ + kc + (t & 3) * 8;
  unsigned vl = (unsigned)(size_t)vdst + (unsigned)t * 16u;
  asm volatile("global_load_async_to_lds_b128 %0, %1, off"
               :: "v"(vl), "v"(vg) : "memory");
}

// ---------------------------------------------------------------------------
// Flash-style attention. Block = 8 waves = 8 consecutive q-tiles of one (b,h);
// K/V chunks staged once per block in LDS (double-buffered async copies).
// Per 32-key iteration: 4 WMMAs for scores, online softmax, LDS transpose of
// P (D-layout -> A-layout), 4 WMMAs for P @ V.
// ---------------------------------------------------------------------------
__global__ __launch_bounds__(256) void attn_kernel(
    const bf16* __restrict__ Qb, const bf16* __restrict__ Kb,
    const bf16* __restrict__ Vt, const float* __restrict__ bias,
    const float* __restrict__ head_mask, bf16* __restrict__ ctx) {
  __shared__ float pbuf[8][16][36];                    // per-wave P tile
  __shared__ __align__(64) bf16 kbuf[2][32][64];       // [key][dim]
  __shared__ __align__(64) bf16 vbuf[2][64][32];       // [dim][key]

  int t    = threadIdx.x;
  int wv   = t >> 5;
  int lane = t & 31;
  int bh   = blockIdx.x >> 4;        // (b*H + h)
  int qblk = blockIdx.x & 15;        // 16 blocks of 128 queries per (b,h)
  int h  = bh & (NH - 1);
  int b  = bh >> 4;
  int q0 = qblk * 128 + wv * 16;

  const bf16* Qh = Qb + (long)bh * SEQ * HD;
  const bf16* Kh = Kb + (long)bh * SEQ * HD;
  const bf16* Vh = Vt + (long)bh * HD * SEQ;
  const float* biasb = bias + (long)b * SEQ * SEQ;

  v16bf aq0 = load_a_frag(Qh, HD, q0, 0, lane);
  v16bf aq1 = load_a_frag(Qh, HD, q0, 32, lane);

  v8f acc0 = {}, acc1 = {}, acc2 = {}, acc3 = {};
  float mi[8], li[8];
#pragma unroll
  for (int r = 0; r < 8; ++r) { mi[r] = -3.0e38f; li[r] = 0.0f; }

  int hi  = lane >> 4;
  int col = lane & 15;
  float (*pl)[36] = pbuf[wv];

  // Preload chunk 0.
  stage_kv(Kh, Vh, 0, &kbuf[0][0][0], &vbuf[0][0][0], t);

  for (int kc = 0; kc < SEQ; kc += 32) {
    int buf = (kc >> 5) & 1;
    // Our own async copies for this chunk are done:
    asm volatile("s_wait_asynccnt 0x0" ::: "memory");
    // All waves' copies landed; also guarantees previous chunk reads finished
    // before the prefetch below overwrites the other buffer.
    __syncthreads();
    if (kc + 32 < SEQ)
      stage_kv(Kh, Vh, kc + 32, &kbuf[buf ^ 1][0][0], &vbuf[buf ^ 1][0][0], t);

    const bf16* kls = &kbuf[buf][0][0];
    const bf16* vls = &vbuf[buf][0][0];

    // ---- scores: two 16-key tiles, K-dim = 64 (two K=32 WMMA chunks) ----
    v8f s0 = {}, s1 = {};
    s0 = wmma_bf16(aq0, load_b_frag(kls, HD, 0, 0, lane), s0);
    s0 = wmma_bf16(aq1, load_b_frag(kls, HD, 0, 32, lane), s0);
    s1 = wmma_bf16(aq0, load_b_frag(kls, HD, 16, 0, lane), s1);
    s1 = wmma_bf16(aq1, load_b_frag(kls, HD, 16, 32, lane), s1);

#pragma unroll
    for (int r = 0; r < 8; ++r) {
      int q = q0 + r + 8 * hi;
      const float* brow = biasb + (long)q * SEQ + kc;
      s0[r] += brow[col];
      s1[r] += brow[16 + col];
    }

    // ---- online softmax (row = 16 lanes within a half-wave) ----
#pragma unroll
    for (int r = 0; r < 8; ++r) {
      float rm = fmaxf(s0[r], s1[r]);
#pragma unroll
      for (int off = 1; off < 16; off <<= 1)
        rm = fmaxf(rm, __shfl_xor(rm, off, 32));
      float nm = fmaxf(mi[r], rm);
      float scale = __expf(mi[r] - nm);
      mi[r] = nm;
      float p0 = __expf(s0[r] - nm);
      float p1 = __expf(s1[r] - nm);
      float rs = p0 + p1;
#pragma unroll
      for (int off = 1; off < 16; off <<= 1)
        rs += __shfl_xor(rs, off, 32);
      li[r] = li[r] * scale + rs;
      acc0[r] *= scale; acc1[r] *= scale; acc2[r] *= scale; acc3[r] *= scale;
      pl[r + 8 * hi][col]      = p0;   // P tile in D-layout
      pl[r + 8 * hi][16 + col] = p1;
    }
    asm volatile("s_wait_dscnt 0" ::: "memory");

    // ---- transpose read: P as a 16x32 bf16 A-fragment ----
    v16bf ap;
    {
      int row = lane & 15;
      int kb  = 8 * hi;
#pragma unroll
      for (int i = 0; i < 8; ++i) {
        ap[i]     = (bf16)pl[row][kb + i];
        ap[i + 8] = (bf16)pl[row][kb + 16 + i];
      }
    }
    // ---- P @ V over Hd = 4 column chunks of 16 (V staged in LDS) ----
    acc0 = wmma_bf16(ap, load_b_frag(vls, 32, 0,  0, lane), acc0);
    acc1 = wmma_bf16(ap, load_b_frag(vls, 32, 16, 0, lane), acc1);
    acc2 = wmma_bf16(ap, load_b_frag(vls, 32, 32, 0, lane), acc2);
    acc3 = wmma_bf16(ap, load_b_frag(vls, 32, 48, 0, lane), acc3);
  }

  // ---- epilogue: /l, *head_mask, store ctx[B,S,D] (bf16) ----
  float hm = head_mask[h];
#pragma unroll
  for (int r = 0; r < 8; ++r) {
    float inv = hm / li[r];
    int q = q0 + r + 8 * hi;
    long orow = ((long)b * SEQ + q) * DIM + h * HD;
    ctx[orow + col]      = (bf16)(acc0[r] * inv);
    ctx[orow + 16 + col] = (bf16)(acc1[r] * inv);
    ctx[orow + 32 + col] = (bf16)(acc2[r] * inv);
    ctx[orow + 48 + col] = (bf16)(acc3[r] * inv);
  }
}

// ---------------------------------------------------------------------------
// Output projection: out[m][n] = sum_k ctx[m][k] * o_w[n][k] + o_b[n]  (f32)
// 32x64 tile per wave.
// ---------------------------------------------------------------------------
__global__ __launch_bounds__(256) void out_gemm_kernel(
    const bf16* __restrict__ ctx, const bf16* __restrict__ owb,
    const float* __restrict__ ob, float* __restrict__ out) {
  int t    = blockIdx.x * 8 + (threadIdx.x >> 5);
  int lane = threadIdx.x & 31;
  int tm = t >> 4;                 // N tiles of 64: 1024/64 = 16
  int tn = t & 15;
  int m0 = tm * 32, n0 = tn * 64;

  v8f acc[2][4] = {};
  for (int kb = 0; kb < DIM; kb += 32) {
    v16bf a0 = load_a_frag(ctx, DIM, m0,      kb, lane);
    v16bf a1 = load_a_frag(ctx, DIM, m0 + 16, kb, lane);
#pragma unroll
    for (int j = 0; j < 4; ++j) {
      v16bf bj = load_b_frag(owb, DIM, n0 + 16 * j, kb, lane);
      acc[0][j] = wmma_bf16(a0, bj, acc[0][j]);
      acc[1][j] = wmma_bf16(a1, bj, acc[1][j]);
    }
  }
  int hi = lane >> 4;
  int cl = lane & 15;
#pragma unroll
  for (int j = 0; j < 4; ++j) {
    int n = n0 + 16 * j + cl;
    float bn = ob[n];
#pragma unroll
    for (int i = 0; i < 2; ++i) {
#pragma unroll
      for (int r = 0; r < 8; ++r) {
        out[(long)(m0 + 16 * i + r + 8 * hi) * DIM + n] = acc[i][j][r] + bn;
      }
    }
  }
}

// ---------------------------------------------------------------------------
// Launch
// ---------------------------------------------------------------------------
extern "C" void kernel_launch(void* const* d_in, const int* in_sizes, int n_in,
                              void* d_out, int out_size, void* d_ws, size_t ws_size,
                              hipStream_t stream) {
  const float* hs     = (const float*)d_in[0];   // [4,2048,1024]
  const float* abias  = (const float*)d_in[1];   // [4,1,2048,2048]
  const float* rcos   = (const float*)d_in[2];   // [1,2048,1,64]
  const float* rsin   = (const float*)d_in[3];   // [1,2048,1,64]
  const float* hmask  = (const float*)d_in[4];   // [1,16,1,1]
  const float* qkv_w  = (const float*)d_in[5];   // [3072,1024]
  const float* qkv_b  = (const float*)d_in[6];   // [3072]
  const float* o_w    = (const float*)d_in[7];   // [1024,1024]
  const float* o_b    = (const float*)d_in[8];   // [1024]
  float* out = (float*)d_out;

  char* ws = (char*)d_ws;
  bf16* hsb   = (bf16*)(ws);                         // 16 MB
  bf16* wqkvb = (bf16*)(ws + 16777216);              //  6 MB
  bf16* owb   = (bf16*)(ws + 23068672);              //  2 MB
  bf16* Qb    = (bf16*)(ws + 25165824);              // 16 MB
  bf16* Kb    = (bf16*)(ws + 41943040);              // 16 MB
  bf16* Vt    = (bf16*)(ws + 58720256);              // 16 MB
  bf16* ctx   = (bf16*)(ws + 75497472);              // 16 MB  (total 88 MB)

  // 1) convert activations + weights to bf16
  cvt_bf16_kernel<<<BATCH * SEQ * DIM / 256, 256, 0, stream>>>(hs, hsb, BATCH * SEQ * DIM);
  cvt_bf16_kernel<<<D3 * DIM / 256, 256, 0, stream>>>(qkv_w, wqkvb, D3 * DIM);
  cvt_bf16_kernel<<<DIM * DIM / 256, 256, 0, stream>>>(o_w, owb, DIM * DIM);

  // 2) QKV projection (+ scatter into Q, K, V^T)
  {
    int tiles = (BATCH * SEQ / 32) * (D3 / 64);      // 12288 waves
    qkv_gemm_kernel<<<tiles / 8, 256, 0, stream>>>(hsb, wqkvb, qkv_b, Qb, Kb, Vt);
  }

  // 3) RoPE on Q, K
  rope_kernel<<<BATCH * NH * SEQ * 32 / 256, 256, 0, stream>>>(Qb, Kb, rcos, rsin);

  // 4) attention (8 q-tiles of one (b,h) per block; async K/V staging)
  attn_kernel<<<BATCH * NH * 16, 256, 0, stream>>>(Qb, Kb, Vt, abias, hmask, ctx);

  // 5) output projection
  {
    int tiles = (BATCH * SEQ / 32) * (DIM / 64);     // 4096 waves
    out_gemm_kernel<<<tiles / 8, 256, 0, stream>>>(ctx, owb, o_b, out);
  }
}
